// F_SelfAttention_867583394562
// MI455X (gfx1250) — matrix operand order repaired and checked by reading, not compiled, not run on hardware
//
#include <hip/hip_runtime.h>
#include <hip/hip_bf16.h>
#include <math.h>

// ---------------- problem constants ----------------
#define TT 8
#define GG 4096
#define SS 256
// elements per batch of the attention matrix
#define NPT (GG * GG)              // 16,777,216
#define SCALE 0.0625f              // 1/sqrt(256)

typedef __attribute__((ext_vector_type(16))) __bf16 v16bf;
typedef __attribute__((ext_vector_type(8)))  __bf16 v8bf;
typedef __attribute__((ext_vector_type(8)))  float  v8f;

static __device__ inline v8f wmma_bf16(v16bf a, v16bf b, v8f c) {
    // D = A(16x32 bf16) x B(32x16 bf16) + C(16x16 f32)
    return __builtin_amdgcn_wmma_f32_16x16x32_bf16(
        /*neg_a=*/false, a, /*neg_b=*/false, b,
        /*c_mod=*/(short)0, c, /*reuse_a=*/false, /*reuse_b=*/false);
}

static __device__ inline v8f vzero8() {
    v8f z = {0.f, 0.f, 0.f, 0.f, 0.f, 0.f, 0.f, 0.f};
    return z;
}

// ---------------- workspace layout (bytes) ----------------
#define OFF_QB   ((size_t)0)                                  // bf16 Q: T*G*S*2 = 16 MB
#define OFF_KB   (OFF_QB + (size_t)TT * GG * SS * 2)          // bf16 K: 16 MB
#define OFF_WQB  (OFF_KB + (size_t)TT * GG * SS * 2)          // bf16 Wq: 128 KB
#define OFF_WKB  (OFF_WQB + (size_t)SS * SS * 2)              // bf16 Wk: 128 KB
#define OFF_RM   (OFF_WKB + (size_t)SS * SS * 2)              // row max: T*G*4
#define OFF_RL   (OFF_RM + (size_t)TT * GG * 4)               // row sum: T*G*4
#define OFF_H1   (OFF_RL + (size_t)TT * GG * 4)               // hist1: T*1024*4
#define OFF_H2   (OFF_H1 + (size_t)TT * 1024 * 4)             // hist2: T*2*4096*4
#define OFF_H3   (OFF_H2 + (size_t)TT * 2 * 4096 * 4)         // hist3: T*2*1024*4
#define OFF_CTL  (OFF_H3 + (size_t)TT * 2 * 1024 * 4)         // ctl:   T*16*4
#define HIST_BYTES ((size_t)TT * 1024 * 4 + (size_t)TT * 2 * 4096 * 4 + (size_t)TT * 2 * 1024 * 4)

// ---------------- kernel 0: weights fp32 -> bf16 ----------------
__global__ void cvt_weights_kernel(const float* __restrict__ qw,
                                   const float* __restrict__ kw,
                                   __bf16* __restrict__ wqb,
                                   __bf16* __restrict__ wkb) {
    int i = blockIdx.x * 256 + threadIdx.x;   // grid covers SS*SS = 65536
    if (i < SS * SS) {
        wqb[i] = (__bf16)qw[i];
        wkb[i] = (__bf16)kw[i];
    }
}

// ---------------- kernel 1: Q/K projection (WMMA bf16) ----------------
// grid = T * (G/64) blocks, 256 threads (8 waves). Each block: 64 rows x 256 cols
// of both Q = x@Wq^T + bq and K = x@Wk^T + bk, outputs bf16.
// 4 accumulator tiles live at a time; ks loop unrolled only 2x to keep at most
// two A fragments live -> no VGPR spills.
__global__ void proj_kernel(const float* __restrict__ x,
                            const __bf16* __restrict__ wqb,
                            const __bf16* __restrict__ wkb,
                            const float* __restrict__ qbias,
                            const float* __restrict__ kbias,
                            __bf16* __restrict__ Qb,
                            __bf16* __restrict__ Kb) {
    const int tid  = threadIdx.x;
    const int w    = tid >> 5;
    const int lane = tid & 31;
    const int lm   = lane & 15;       // A: M index / B: N index / D: N index
    const int half = lane >> 4;
    const int t    = blockIdx.x >> 6;
    const int g0   = (blockIdx.x & 63) * 64;
    const int r    = w & 3;           // row tile (16 rows)
    const int ch   = w >> 2;          // column half (8 col tiles each)

    const int arow = g0 + r * 16 + lm;
    const float* xrow = x + ((size_t)(t * GG + arow)) * SS;

#pragma unroll 1
    for (int p = 0; p < 2; ++p) {
        const __bf16* W    = p ? wkb : wqb;
        const float*  bias = p ? kbias : qbias;
        __bf16*       Out  = p ? Kb : Qb;

#pragma unroll 1
        for (int cp = 0; cp < 2; ++cp) {      // two passes of 4 col tiles
            const int ct0 = ch * 8 + cp * 4;  // first of 4 col tiles this pass

            v8f acc[4];
#pragma unroll
            for (int i = 0; i < 4; ++i) acc[i] = vzero8();

#pragma unroll 2
            for (int ks = 0; ks < 8; ++ks) {
                // A fragment: two contiguous runs of 8 fp32 -> bf16
                const float* p0 = xrow + ks * 32 + half * 16;
                v16bf a;
#pragma unroll
                for (int i = 0; i < 8; ++i) {
                    a[i]     = (__bf16)p0[i];
                    a[i + 8] = (__bf16)p0[8 + i];
                }
#pragma unroll
                for (int c4 = 0; c4 < 4; ++c4) {
                    const int n = (ct0 + c4) * 16 + lm;
                    v16bf b = *(const v16bf*)(W + n * SS + ks * 32 + half * 16);
                    acc[c4] = wmma_bf16(a, b, acc[c4]);
                }
            }
            // bias add on D layout + bf16 store
#pragma unroll
            for (int c4 = 0; c4 < 4; ++c4) {
                const int n  = (ct0 + c4) * 16 + lm;
                const float bn = bias[n];
#pragma unroll
                for (int v = 0; v < 8; ++v) {
                    const int grow = g0 + r * 16 + half * 8 + v;
                    Out[((size_t)(t * GG + grow)) * SS + n] = (__bf16)(acc[c4][v] + bn);
                }
            }
        }
    }
}

// ---------------- kernel 2: scores + online softmax stats ----------------
// grid = T * (G/64), 256 threads. Wave owns 16 query rows; streams all 256 key
// tiles (two waves interleave the column space), writes raw scaled scores and
// produces exact per-row (max m, sum l = sum exp(s-m)).
__global__ void scores_kernel(const __bf16* __restrict__ Qb,
                              const __bf16* __restrict__ Kb,
                              float* __restrict__ out,
                              float* __restrict__ row_m,
                              float* __restrict__ row_l) {
    const int tid  = threadIdx.x;
    const int w    = tid >> 5;
    const int lane = tid & 31;
    const int lm   = lane & 15;
    const int half = lane >> 4;
    const int t    = blockIdx.x >> 6;
    const int g0   = (blockIdx.x & 63) * 64;
    const int r    = w & 3;

    // preload all 8 A k-fragments for this wave's 16 rows (bf16)
    const __bf16* qrow = Qb + ((size_t)(t * GG + g0 + r * 16 + lm)) * SS;
    v16bf aq[8];
#pragma unroll
    for (int ks = 0; ks < 8; ++ks) {
        v8bf lo = *(const v8bf*)(qrow + ks * 32 + half * 16);
        v8bf hi = *(const v8bf*)(qrow + ks * 32 + half * 16 + 8);
#pragma unroll
        for (int i = 0; i < 8; ++i) { aq[ks][i] = lo[i]; aq[ks][i + 8] = hi[i]; }
    }

    float m[8], l[8];
#pragma unroll
    for (int v = 0; v < 8; ++v) { m[v] = -__builtin_inff(); l[v] = 0.f; }

#pragma unroll 1
    for (int ct = (w >> 2); ct < GG / 16; ct += 2) {
        const __bf16* kbase = Kb + ((size_t)(t * GG + ct * 16 + lm)) * SS + half * 16;
        if (ct + 2 < GG / 16)
            __builtin_prefetch(kbase + 2 * 16 * SS, 0, 0);   // global_prefetch next tile

        v8f acc = vzero8();
#pragma unroll
        for (int ks = 0; ks < 8; ++ks) {
            v16bf b = *(const v16bf*)(kbase + ks * 32);
            acc = wmma_bf16(aq[ks], b, acc);
        }

#pragma unroll
        for (int v = 0; v < 8; ++v) {
            const float s = acc[v] * SCALE;
            const int grow = g0 + r * 16 + half * 8 + v;
            out[(((size_t)t * GG + grow) << 12) + ct * 16 + lm] = s;

            // row max over the 16 lanes of this half (xor 1,2,4,8 stays in half)
            float mx = s;
            mx = fmaxf(mx, __shfl_xor(mx, 1));
            mx = fmaxf(mx, __shfl_xor(mx, 2));
            mx = fmaxf(mx, __shfl_xor(mx, 4));
            mx = fmaxf(mx, __shfl_xor(mx, 8));
            const float mn = fmaxf(m[v], mx);
            float ps = expf(s - mn);
            ps += __shfl_xor(ps, 1);
            ps += __shfl_xor(ps, 2);
            ps += __shfl_xor(ps, 4);
            ps += __shfl_xor(ps, 8);
            l[v] = l[v] * expf(m[v] - mn) + ps;
            m[v] = mn;
        }
    }

    // combine the two waves (w and w+4) sharing the same 16 rows
    __shared__ float sm[8][16];
    __shared__ float sl[8][16];
    if (lm == 0) {
#pragma unroll
        for (int v = 0; v < 8; ++v) {
            const int rl = half * 8 + v;
            sm[w][rl] = m[v];
            sl[w][rl] = l[v];
        }
    }
    __syncthreads();
    if (w < 4 && lm == 0) {
#pragma unroll
        for (int v = 0; v < 8; ++v) {
            const int rl = half * 8 + v;
            const float m1 = sm[w][rl], l1 = sl[w][rl];
            const float m2 = sm[w + 4][rl], l2 = sl[w + 4][rl];
            const float M = fmaxf(m1, m2);
            const float L = l1 * expf(m1 - M) + l2 * expf(m2 - M);
            const int grow = g0 + r * 16 + rl;
            row_m[t * GG + grow] = M;
            row_l[t * GG + grow] = L;
        }
    }
}

// ---------------- kernel 3: softmax normalize + level-1 histogram ----------------
// grid = T*1024 blocks, 256 threads, 16384 elems/block (float4).
__global__ void norm_hist_kernel(float* __restrict__ out,
                                 const float* __restrict__ row_m,
                                 const float* __restrict__ row_l,
                                 unsigned int* __restrict__ hist1) {
    __shared__ unsigned int h[1024];
    const int tid = threadIdx.x;
    for (int i = tid; i < 1024; i += 256) h[i] = 0;
    __syncthreads();

    const int t = blockIdx.x >> 10;
    const int chunk = (blockIdx.x & 1023) * 16384;
    const size_t tbase = (size_t)t << 24;   // t * G*G

#pragma unroll 4
    for (int it = 0; it < 16; ++it) {
        const int idx = chunk + it * 1024 + tid * 4;
        float4 v = *(float4*)(out + tbase + idx);
        const int row = idx >> 12;
        const float mm = row_m[t * GG + row];
        const float inv = 1.0f / row_l[t * GG + row];
        float4 a;
        a.x = expf(v.x - mm) * inv;
        a.y = expf(v.y - mm) * inv;
        a.z = expf(v.z - mm) * inv;
        a.w = expf(v.w - mm) * inv;
        *(float4*)(out + tbase + idx) = a;
        atomicAdd(&h[__float_as_uint(a.x) >> 22], 1u);
        atomicAdd(&h[__float_as_uint(a.y) >> 22], 1u);
        atomicAdd(&h[__float_as_uint(a.z) >> 22], 1u);
        atomicAdd(&h[__float_as_uint(a.w) >> 22], 1u);
    }
    __syncthreads();
    for (int i = tid; i < 1024; i += 256)
        if (h[i]) atomicAdd(&hist1[t * 1024 + i], h[i]);
}

// ---------------- radix-select scan helpers ----------------
__global__ void select1_kernel(const unsigned int* __restrict__ hist1,
                               unsigned int* __restrict__ ctl) {
    if (threadIdx.x != 0) return;
    const int t = blockIdx.x;
    const double pos = 0.98 * (double)(NPT - 1);
    unsigned long long ranks[2];
    ranks[0] = (unsigned long long)pos;       // k
    ranks[1] = ranks[0] + 1ull;               // k+1
    for (int s = 0; s < 2; ++s) {
        unsigned long long cum = 0;
        unsigned int bin = 1023, rem = 0;
        for (int b = 0; b < 1024; ++b) {
            const unsigned int c = hist1[t * 1024 + b];
            if (cum + c > ranks[s]) { bin = (unsigned)b; rem = (unsigned)(ranks[s] - cum); break; }
            cum += c;
        }
        ctl[t * 16 + s * 2 + 0] = bin;   // 10-bit prefix (bits 31:22)
        ctl[t * 16 + s * 2 + 1] = rem;   // remaining rank within bin
    }
}

__global__ void hist2_kernel(const float* __restrict__ out,
                             const unsigned int* __restrict__ ctl,
                             unsigned int* __restrict__ hist2) {
    __shared__ unsigned int h[2][4096];
    const int tid = threadIdx.x;
    for (int i = tid; i < 8192; i += 256) h[i >> 12][i & 4095] = 0;
    __syncthreads();

    const int t = blockIdx.x >> 10;
    const unsigned int pA = ctl[t * 16 + 0];
    const unsigned int pB = ctl[t * 16 + 2];
    const int chunk = (blockIdx.x & 1023) * 16384;
    const size_t tbase = (size_t)t << 24;

#pragma unroll 4
    for (int it = 0; it < 16; ++it) {
        const int idx = chunk + it * 1024 + tid * 4;
        float4 v = *(const float4*)(out + tbase + idx);
        const unsigned int bb[4] = {__float_as_uint(v.x), __float_as_uint(v.y),
                                    __float_as_uint(v.z), __float_as_uint(v.w)};
#pragma unroll
        for (int j = 0; j < 4; ++j) {
            const unsigned int pr = bb[j] >> 22;
            if (pr == pA) atomicAdd(&h[0][(bb[j] >> 10) & 4095], 1u);
            if (pr == pB) atomicAdd(&h[1][(bb[j] >> 10) & 4095], 1u);
        }
    }
    __syncthreads();
    for (int i = tid; i < 8192; i += 256) {
        const unsigned int c = h[i >> 12][i & 4095];
        if (c) atomicAdd(&hist2[t * 8192 + i], c);
    }
}

__global__ void select2_kernel(const unsigned int* __restrict__ hist2,
                               unsigned int* __restrict__ ctl) {
    if (threadIdx.x != 0) return;
    const int t = blockIdx.x;
    for (int s = 0; s < 2; ++s) {
        const unsigned int pref = ctl[t * 16 + s * 2 + 0];
        const unsigned long long rank = ctl[t * 16 + s * 2 + 1];
        unsigned long long cum = 0;
        unsigned int bin = 4095, rem = 0;
        for (int b = 0; b < 4096; ++b) {
            const unsigned int c = hist2[t * 8192 + s * 4096 + b];
            if (cum + c > rank) { bin = (unsigned)b; rem = (unsigned)(rank - cum); break; }
            cum += c;
        }
        ctl[t * 16 + s * 2 + 0] = (pref << 12) | bin;  // 22-bit prefix (bits 31:10)
        ctl[t * 16 + s * 2 + 1] = rem;
    }
}

__global__ void hist3_kernel(const float* __restrict__ out,
                             const unsigned int* __restrict__ ctl,
                             unsigned int* __restrict__ hist3) {
    __shared__ unsigned int h[2][1024];
    const int tid = threadIdx.x;
    for (int i = tid; i < 2048; i += 256) h[i >> 10][i & 1023] = 0;
    __syncthreads();

    const int t = blockIdx.x >> 10;
    const unsigned int pA = ctl[t * 16 + 0];
    const unsigned int pB = ctl[t * 16 + 2];
    const int chunk = (blockIdx.x & 1023) * 16384;
    const size_t tbase = (size_t)t << 24;

#pragma unroll 4
    for (int it = 0; it < 16; ++it) {
        const int idx = chunk + it * 1024 + tid * 4;
        float4 v = *(const float4*)(out + tbase + idx);
        const unsigned int bb[4] = {__float_as_uint(v.x), __float_as_uint(v.y),
                                    __float_as_uint(v.z), __float_as_uint(v.w)};
#pragma unroll
        for (int j = 0; j < 4; ++j) {
            const unsigned int pr = bb[j] >> 10;
            if (pr == pA) atomicAdd(&h[0][bb[j] & 1023], 1u);
            if (pr == pB) atomicAdd(&h[1][bb[j] & 1023], 1u);
        }
    }
    __syncthreads();
    for (int i = tid; i < 2048; i += 256) {
        const unsigned int c = h[i >> 10][i & 1023];
        if (c) atomicAdd(&hist3[t * 2048 + i], c);
    }
}

__global__ void select3_kernel(const unsigned int* __restrict__ hist3,
                               unsigned int* __restrict__ ctl) {
    if (threadIdx.x != 0) return;
    const int t = blockIdx.x;
    unsigned int vbits[2];
    for (int s = 0; s < 2; ++s) {
        const unsigned int pref = ctl[t * 16 + s * 2 + 0];
        const unsigned long long rank = ctl[t * 16 + s * 2 + 1];
        unsigned long long cum = 0;
        unsigned int bin = 1023;
        for (int b = 0; b < 1024; ++b) {
            const unsigned int c = hist3[t * 2048 + s * 1024 + b];
            if (cum + c > rank) { bin = (unsigned)b; break; }
            cum += c;
        }
        vbits[s] = (pref << 10) | bin;   // full 32-bit pattern of order stat
    }
    const double pos = 0.98 * (double)(NPT - 1);
    const double frac = pos - (double)(unsigned long long)pos;
    const float vlo = __uint_as_float(vbits[0]);
    const float vhi = __uint_as_float(vbits[1]);
    const float thr = vlo + (float)frac * (vhi - vlo);
    ctl[t * 16 + 6] = __float_as_uint(thr);
}

// ---------------- kernel 9: relu(att - thr) ----------------
__global__ void relu_thresh_kernel(float* __restrict__ out,
                                   const unsigned int* __restrict__ ctl) {
    const int tid = threadIdx.x;
    const int t = blockIdx.x >> 10;
    const float thr = __uint_as_float(ctl[t * 16 + 6]);
    const int chunk = (blockIdx.x & 1023) * 16384;
    const size_t tbase = (size_t)t << 24;
#pragma unroll 4
    for (int it = 0; it < 16; ++it) {
        const int idx = chunk + it * 1024 + tid * 4;
        float4 v = *(float4*)(out + tbase + idx);
        v.x = fmaxf(v.x - thr, 0.f);
        v.y = fmaxf(v.y - thr, 0.f);
        v.z = fmaxf(v.z - thr, 0.f);
        v.w = fmaxf(v.w - thr, 0.f);
        *(float4*)(out + tbase + idx) = v;
    }
}

// ---------------- host launcher ----------------
extern "C" void kernel_launch(void* const* d_in, const int* in_sizes, int n_in,
                              void* d_out, int out_size, void* d_ws, size_t ws_size,
                              hipStream_t stream) {
    const float* x  = (const float*)d_in[0];
    const float* qw = (const float*)d_in[1];
    const float* kw = (const float*)d_in[2];
    const float* qb = (const float*)d_in[3];
    const float* kb = (const float*)d_in[4];
    float* out = (float*)d_out;

    char* ws = (char*)d_ws;
    __bf16* Qb  = (__bf16*)(ws + OFF_QB);
    __bf16* Kb  = (__bf16*)(ws + OFF_KB);
    __bf16* Wqb = (__bf16*)(ws + OFF_WQB);
    __bf16* Wkb = (__bf16*)(ws + OFF_WKB);
    float* rm = (float*)(ws + OFF_RM);
    float* rl = (float*)(ws + OFF_RL);
    unsigned int* h1  = (unsigned int*)(ws + OFF_H1);
    unsigned int* h2  = (unsigned int*)(ws + OFF_H2);
    unsigned int* h3  = (unsigned int*)(ws + OFF_H3);
    unsigned int* ctl = (unsigned int*)(ws + OFF_CTL);

    cvt_weights_kernel<<<dim3(SS * SS / 256), dim3(256), 0, stream>>>(qw, kw, Wqb, Wkb);
    proj_kernel<<<dim3(TT * (GG / 64)), dim3(256), 0, stream>>>(x, Wqb, Wkb, qb, kb, Qb, Kb);
    scores_kernel<<<dim3(TT * (GG / 64)), dim3(256), 0, stream>>>(Qb, Kb, out, rm, rl);

    hipMemsetAsync(h1, 0, HIST_BYTES, stream);   // h1/h2/h3 are contiguous
    norm_hist_kernel<<<dim3(TT * 1024), dim3(256), 0, stream>>>(out, rm, rl, h1);
    select1_kernel<<<dim3(TT), dim3(32), 0, stream>>>(h1, ctl);
    hist2_kernel<<<dim3(TT * 1024), dim3(256), 0, stream>>>(out, ctl, h2);
    select2_kernel<<<dim3(TT), dim3(32), 0, stream>>>(h2, ctl);
    hist3_kernel<<<dim3(TT * 1024), dim3(256), 0, stream>>>(out, ctl, h3);
    select3_kernel<<<dim3(TT), dim3(32), 0, stream>>>(h3, ctl);
    relu_thresh_kernel<<<dim3(TT * 1024), dim3(256), 0, stream>>>(out, ctl);
}
